// GCNB_59107339927855
// MI455X (gfx1250) — matrix operand order repaired and checked by reading, not compiled, not run on hardware
//
#include <hip/hip_runtime.h>
#include <hip/hip_bf16.h>
#include <math.h>

// Problem constants (match reference)
#define NN   8192      // nodes
#define FDIM 128       // feature dim
#define BATCH 4
#define COLS (BATCH * FDIM)   // 512 combined (b,e) columns for the big GEMM

typedef __attribute__((ext_vector_type(16))) __bf16       v16bf;
typedef __attribute__((ext_vector_type(8)))  float        v8f;
typedef __attribute__((ext_vector_type(4)))  unsigned int v4u;

struct Frag32B { v4u lo; v4u hi; };

__device__ __forceinline__ v16bf load_frag(const unsigned short* p0,
                                           const unsigned short* p1) {
    Frag32B t;
    t.lo = *(const v4u*)p0;   // 16 bytes (8 bf16)
    t.hi = *(const v4u*)p1;   // 16 bytes (8 bf16)
    return __builtin_bit_cast(v16bf, t);
}

__device__ __forceinline__ unsigned short f2bf(float f) {
    unsigned u = __builtin_bit_cast(unsigned, f);
    unsigned r = u + 0x7FFFu + ((u >> 16) & 1u);   // round-to-nearest-even
    return (unsigned short)(r >> 16);
}
__device__ __forceinline__ float bf2f(unsigned short s) {
    return __builtin_bit_cast(float, (unsigned)s << 16);
}
__device__ __forceinline__ float leaky(float v) { return v >= 0.f ? v : 0.01f * v; }
__device__ __forceinline__ float mishf(float y) {
    float sp = (y > 20.f) ? y : log1pf(__expf(y));   // softplus, overflow-safe
    return y * tanhf(sp);
}

// ---------------------------------------------------------------------------
// deg[n] = sum_m g[n,m];  rsq[n] = 1/sqrt(deg[n])
// ---------------------------------------------------------------------------
__global__ void deg_rsqrt_kernel(const float* __restrict__ g, float* __restrict__ rsq) {
    __shared__ float red[256];
    int n = blockIdx.x;
    const float* row = g + (size_t)n * NN;
    float s = 0.f;
    for (int m = threadIdx.x; m < NN; m += 256) s += row[m];
    red[threadIdx.x] = s;
    __syncthreads();
    for (int o = 128; o > 0; o >>= 1) {
        if ((int)threadIdx.x < o) red[threadIdx.x] += red[threadIdx.x + o];
        __syncthreads();
    }
    if (threadIdx.x == 0) rsq[n] = rsqrtf(red[0]);
}

// ---------------------------------------------------------------------------
// Ab[n,m] = bf16( g[n,m] * rsq[n] * rsq[m] )   (128 MB -> L2 resident)
// ---------------------------------------------------------------------------
__global__ void normadj_kernel(const float* __restrict__ g, const float* __restrict__ rsq,
                               unsigned short* __restrict__ Ab) {
    const size_t total = (size_t)NN * NN;
    size_t stride = (size_t)gridDim.x * blockDim.x;
    for (size_t i = (size_t)blockIdx.x * blockDim.x + threadIdx.x; i < total; i += stride) {
        int n = (int)(i >> 13);
        int m = (int)(i & (NN - 1));
        Ab[i] = f2bf(g[i] * rsq[n] * rsq[m]);
    }
}

// ---------------------------------------------------------------------------
// X[b,n,e] = leaky(data[b,n,:3] @ w_lin1 + b_lin1) + sincos_emb(t)[e]
// grid = B*NN blocks, block = 128 threads (thread = e)
// ---------------------------------------------------------------------------
__global__ void init_x_kernel(const float* __restrict__ data, const float* __restrict__ w1,
                              const float* __restrict__ b1, const int* __restrict__ tptr,
                              float* __restrict__ X) {
    int bn = blockIdx.x;
    int e  = threadIdx.x;
    const float* d = data + (size_t)bn * 3;
    float acc = b1[e] + d[0] * w1[0 * FDIM + e] + d[1] * w1[1 * FDIM + e]
                       + d[2] * w1[2 * FDIM + e];
    acc = leaky(acc);
    int   t  = *tptr;
    int   i  = e >> 1;
    float om = __expf(-logf(10000.f) * (float)i * (1.0f / 64.0f));
    float ph = (float)t * om;
    acc += (e & 1) ? cosf(ph) : sinf(ph);
    X[(size_t)bn * FDIM + e] = acc;
}

// ---------------------------------------------------------------------------
// Xt[col, m] = bf16( H[b,m,:] @ W[:,e] )  with col = b*128+e  (transposed store
// so the WMMA B-fragment reads are contiguous in K)
// grid = B*NN blocks (one (b,m) row), block = 128 threads (thread = e)
// ---------------------------------------------------------------------------
__global__ void xwt_kernel(const float* __restrict__ Hin, const float* __restrict__ W,
                           unsigned short* __restrict__ Xt) {
    __shared__ float hrow[FDIM];
    int bm = blockIdx.x;
    int e  = threadIdx.x;
    hrow[e] = Hin[(size_t)bm * FDIM + e];
    __syncthreads();
    float acc = 0.f;
#pragma unroll 8
    for (int d = 0; d < FDIM; ++d) acc += hrow[d] * W[d * FDIM + e];
    int b = bm / NN, m = bm - b * NN;
    int col = b * FDIM + e;
    Xt[(size_t)col * NN + m] = f2bf(acc);
}

// ---------------------------------------------------------------------------
// Big GEMM: C[b,m,e] = leaky( sum_k Ab[m,k]*Xt[col,k] + Xt[col,m] + bias[e] )
// block = 128 threads (4 waves); wave w owns a 32-row x 64-col tile:
// rows [bx*128 + 32w, +32) as two 16-row A fragments sharing each B fragment.
// Per K-step: 4 A loads + 8 B loads feed 8 v_wmma_f32_16x16x32_bf16
// (1.5 b128 loads per WMMA).  Fragment loads follow the ISA 16-bit layouts:
//   A 16x32: lanes 0-15 row M=l, K in {0..7,16..23}; lanes 16-31 K in {8..15,24..31}
//   B 32x16: lanes 0-15 col N=l, K in {0..7,16..23}; lanes 16-31 K in {8..15,24..31} (same as A)
// ---------------------------------------------------------------------------
__global__ __launch_bounds__(128)
void gemm_wmma_kernel(const unsigned short* __restrict__ Ab,
                      const unsigned short* __restrict__ Xt,
                      const float* __restrict__ bias,
                      float* __restrict__ Cout) {
    int tid  = threadIdx.x;
    int wave = tid >> 5;
    int lane = tid & 31;
    int lr   = lane & 15;
    int hi   = lane >> 4;
    int rowBase = blockIdx.x * 128 + wave * 32;
    int colBase = blockIdx.y * 64;

    v8f acc[2][4] = {};

    const unsigned short* arow0 = Ab + (size_t)(rowBase + lr) * NN;
    const unsigned short* arow1 = arow0 + (size_t)16 * NN;

    for (int k0 = 0; k0 < NN; k0 += 32) {
        v16bf a0 = load_frag(arow0 + k0 + hi * 8, arow0 + k0 + 16 + hi * 8);
        v16bf a1 = load_frag(arow1 + k0 + hi * 8, arow1 + k0 + 16 + hi * 8);
        // Pull the adjacency rows ~2KB ahead into near caches
        // (global_prefetch_b8; helps conv1 when Ab is not yet L2-resident).
        __builtin_prefetch(arow0 + k0 + 1024, 0, 3);
        __builtin_prefetch(arow1 + k0 + 1024, 0, 3);
#pragma unroll
        for (int ct = 0; ct < 4; ++ct) {
            int col = colBase + ct * 16 + lr;
            const unsigned short* bp = Xt + (size_t)col * NN + k0;
            v16bf bfrag = load_frag(bp + hi * 8, bp + 16 + hi * 8);
            acc[0][ct] = __builtin_amdgcn_wmma_f32_16x16x32_bf16(
                false, a0, false, bfrag, (short)0, acc[0][ct], false, false);
            acc[1][ct] = __builtin_amdgcn_wmma_f32_16x16x32_bf16(
                false, a1, false, bfrag, (short)0, acc[1][ct], false, false);
        }
    }

    // Epilogue: + XW + bias, LeakyReLU, scatter to [B, N, 128] fp32.
    // C/D layout: lanes 0-15 -> M = v, lanes 16-31 -> M = v + 8, N = lane&15.
#pragma unroll
    for (int rt = 0; rt < 2; ++rt) {
#pragma unroll
        for (int ct = 0; ct < 4; ++ct) {
            int col = colBase + ct * 16 + lr;
            int b = col >> 7;
            int e = col & (FDIM - 1);
            float be = bias[e];
#pragma unroll
            for (int v = 0; v < 8; ++v) {
                int m = rowBase + rt * 16 + v + 8 * hi;
                float val = acc[rt][ct][v] + bf2f(Xt[(size_t)col * NN + m]) + be;
                Cout[((size_t)b * NN + m) * FDIM + e] = leaky(val);
            }
        }
    }
}

// ---------------------------------------------------------------------------
// H[b,n,e] = mish( BN(C)[b,n,e] ) + Res[b,n,e]
// BatchNorm1d(N): stats per node n over (b, e) -> 512 values.
// grid = NN blocks (node), block = 128 threads (thread = e, loops b).
// ---------------------------------------------------------------------------
__global__ void bn_mish_res_kernel(const float* __restrict__ Cin,
                                   const float* __restrict__ gamma,
                                   const float* __restrict__ beta,
                                   const float* __restrict__ Res,
                                   float* __restrict__ Hout) {
    __shared__ float rs[128], rs2[128];
    int n = blockIdx.x;
    int t = threadIdx.x;
    float vals[BATCH];
    float s = 0.f, s2 = 0.f;
#pragma unroll
    for (int b = 0; b < BATCH; ++b) {
        float v = Cin[((size_t)b * NN + n) * FDIM + t];
        vals[b] = v;
        s += v;
        s2 += v * v;
    }
    rs[t] = s; rs2[t] = s2;
    __syncthreads();
    for (int o = 64; o > 0; o >>= 1) {
        if (t < o) { rs[t] += rs[t + o]; rs2[t] += rs2[t + o]; }
        __syncthreads();
    }
    float mean = rs[0] * (1.0f / 512.0f);
    float var  = rs2[0] * (1.0f / 512.0f) - mean * mean;
    float inv  = rsqrtf(var + 1e-5f);
    float ga = gamma[n], be = beta[n];
#pragma unroll
    for (int b = 0; b < BATCH; ++b) {
        float y = ga * ((vals[b] - mean) * inv) + be;
        size_t idx = ((size_t)b * NN + n) * FDIM + t;
        Hout[idx] = mishf(y) + Res[idx];
    }
}

// ---------------------------------------------------------------------------
// Head: out[b,n] = sigmoid( relu(H @ w2 + b2) @ w3 + b3 )
// block = 256 threads = 8 waves; each wave owns one (b,n); lane = feature f.
// ---------------------------------------------------------------------------
__global__ void head_kernel(const float* __restrict__ H,
                            const float* __restrict__ w2, const float* __restrict__ b2,
                            const float* __restrict__ w3, const float* __restrict__ b3,
                            float* __restrict__ out) {
    int wave = threadIdx.x >> 5;
    int lane = threadIdx.x & 31;
    int bn = blockIdx.x * 8 + wave;
    const float* hrow = H + (size_t)bn * FDIM;
    float acc = b2[lane];
#pragma unroll 8
    for (int d = 0; d < FDIM; ++d) acc += hrow[d] * w2[d * 32 + lane];
    acc = fmaxf(acc, 0.f);
    float p = acc * w3[lane];
#pragma unroll
    for (int o = 16; o > 0; o >>= 1) p += __shfl_xor(p, o, 32);
    if (lane == 0) out[bn] = 1.0f / (1.0f + __expf(-(p + b3[0])));
}

// ---------------------------------------------------------------------------
extern "C" void kernel_launch(void* const* d_in, const int* in_sizes, int n_in,
                              void* d_out, int out_size, void* d_ws, size_t ws_size,
                              hipStream_t stream) {
    (void)in_sizes; (void)n_in; (void)out_size; (void)ws_size;
    const float* g      = (const float*)d_in[0];
    const float* data   = (const float*)d_in[1];
    const int*   t      = (const int*)  d_in[2];
    const float* w_lin1 = (const float*)d_in[3];
    const float* b_lin1 = (const float*)d_in[4];
    const float* W1     = (const float*)d_in[5];
    const float* b1     = (const float*)d_in[6];
    const float* W2     = (const float*)d_in[7];
    const float* b2     = (const float*)d_in[8];
    const float* W3     = (const float*)d_in[9];
    const float* b3     = (const float*)d_in[10];
    const float* gamma  = (const float*)d_in[11];
    const float* beta   = (const float*)d_in[12];
    const float* w_lin2 = (const float*)d_in[13];
    const float* b_lin2 = (const float*)d_in[14];
    const float* w_lin3 = (const float*)d_in[15];
    const float* b_lin3 = (const float*)d_in[16];
    float* out = (float*)d_out;

    // Workspace carve-out (~185 MB)
    char* ws = (char*)d_ws;
    unsigned short* Ab = (unsigned short*)ws;  ws += (size_t)NN * NN * sizeof(unsigned short);
    float* rsq = (float*)ws;                   ws += (size_t)NN * sizeof(float);
    float* X   = (float*)ws;                   ws += (size_t)BATCH * NN * FDIM * sizeof(float);
    float* H   = (float*)ws;                   ws += (size_t)BATCH * NN * FDIM * sizeof(float);
    float* C   = (float*)ws;                   ws += (size_t)BATCH * NN * FDIM * sizeof(float);
    unsigned short* Xt = (unsigned short*)ws;  ws += (size_t)COLS * NN * sizeof(unsigned short);

    // One HBM pass over g; Ab (128 MB bf16) stays L2-resident for all 7 convs.
    deg_rsqrt_kernel<<<NN, 256, 0, stream>>>(g, rsq);
    normadj_kernel<<<8192, 256, 0, stream>>>(g, rsq, Ab);
    init_x_kernel<<<BATCH * NN, FDIM, 0, stream>>>(data, w_lin1, b_lin1, t, X);

    auto conv = [&](const float* in, const float* W, const float* bb, float* outH) {
        xwt_kernel<<<BATCH * NN, FDIM, 0, stream>>>(in, W, Xt);
        gemm_wmma_kernel<<<dim3(NN / 128, COLS / 64), 128, 0, stream>>>(Ab, Xt, bb, C);
        bn_mish_res_kernel<<<NN, FDIM, 0, stream>>>(C, gamma, beta, in, outH);
    };

    conv(X, W1, b1, H);                            // conv1 (+ residual x)
    for (int i = 0; i < 5; ++i) conv(H, W2, b2, H); // conv2 applied 5x (shared W)
    conv(H, W3, b3, H);                            // conv3

    head_kernel<<<(BATCH * NN) / 8, 256, 0, stream>>>(H, w_lin2, b_lin2, w_lin3, b_lin3, out);
}
